// Motify_Classifier_49151605735713
// MI455X (gfx1250) — matrix operand order repaired and checked by reading, not compiled, not run on hardware
//
#include <hip/hip_runtime.h>
#include <hip/hip_bf16.h>

#define N_NODES 8192
#define EMB 300
#define N_EDGES 131072
#define LDZ 320          // K padded to 320 = 10 * 32
#define MASK_WORDS (N_NODES * 256)  // 8192 rows * 8192 bits

typedef __attribute__((ext_vector_type(16))) __bf16 v16bf;
typedef __attribute__((ext_vector_type(8)))  __bf16 v8bf;
typedef __attribute__((ext_vector_type(8)))  float  v8f;

// ---------------- kernel 1: zero the adjacency bitmask ----------------
__global__ void zero_mask_kernel(uint4* mask4) {
    int gid = blockIdx.x * blockDim.x + threadIdx.x;   // 2048*256 = 524288 -> *4 words
    uint4 z; z.x = 0u; z.y = 0u; z.z = 0u; z.w = 0u;
    mask4[gid] = z;
}

// ---------------- kernel 2: scatter edges (symmetric, atomicOr dedups) ----------------
__global__ void edge_kernel(const int* __restrict__ ei, unsigned* __restrict__ mask) {
    int e = blockIdx.x * blockDim.x + threadIdx.x;
    if (e >= N_EDGES) return;
    int u = ei[e];
    int v = ei[N_EDGES + e];
    atomicOr(&mask[u * 256 + (v >> 5)], 1u << (v & 31));
    atomicOr(&mask[v * 256 + (u >> 5)], 1u << (u & 31));
}

// ---------------- kernel 3: Hm = x @ Wm, Hl = x @ Wl (f32, 8-row blocking) ----------------
__global__ void xw_kernel(const float* __restrict__ x,
                          const float* __restrict__ Wm,
                          const float* __restrict__ Wl,
                          float* __restrict__ Hm,
                          float* __restrict__ Hl) {
    int j  = blockIdx.x * 128 + threadIdx.x;   // column of W
    int i0 = blockIdx.y * 8;                   // 8 rows of x per block
    if (j >= EMB) return;
    float am[8], al[8];
#pragma unroll
    for (int r = 0; r < 8; ++r) { am[r] = 0.f; al[r] = 0.f; }
    for (int k = 0; k < EMB; ++k) {
        float wm = Wm[k * EMB + j];
        float wl = Wl[k * EMB + j];
#pragma unroll
        for (int r = 0; r < 8; ++r) {
            float xv = x[(i0 + r) * EMB + k];
            am[r] += xv * wm;
            al[r] += xv * wl;
        }
    }
#pragma unroll
    for (int r = 0; r < 8; ++r) {
        Hm[(i0 + r) * EMB + j] = am[r];
        Hl[(i0 + r) * EMB + j] = al[r];
    }
}

// ---------------- kernel 4: sparse aggregate + reparameterize -> Z (bf16, K padded) ----------------
__global__ void agg_kernel(const unsigned* __restrict__ mask,
                           const float* __restrict__ Hm,
                           const float* __restrict__ Hl,
                           const float* __restrict__ noise,
                           __bf16* __restrict__ Z) {
    __shared__ unsigned short nbr[N_NODES];
    __shared__ int cnt;
    int u = blockIdx.x;
    int t = threadIdx.x;          // 0..319 (10 waves)
    if (t == 0) cnt = 0;
    __syncthreads();
    if (t < 256) {
        unsigned w = mask[u * 256 + t];
        while (w) {
            int b = __builtin_ctz(w);
            w &= w - 1u;
            int pos = atomicAdd(&cnt, 1);
            nbr[pos] = (unsigned short)(t * 32 + b);
        }
    }
    __syncthreads();
    int n = cnt;
    if (t < EMB) {
        float am = 0.f, al = 0.f;
        for (int p = 0; p < n; ++p) {
            int v = nbr[p];                       // LDS broadcast
            am += Hm[v * EMB + t];                // coalesced, L2-resident
            al += Hl[v * EMB + t];
        }
        float z = noise[u * EMB + t] * __expf(al) + am;
        Z[u * LDZ + t] = (__bf16)z;
    } else if (t < LDZ) {
        Z[u * LDZ + t] = (__bf16)0.0f;            // zero pad K=300..319
    }
}

// ---------------- kernel 5: out = sigmoid(Z @ Z^T), bf16 WMMA, 32x32 tile per wave ----------------
__device__ __forceinline__ float sigmoidf(float x) {
    return 1.0f / (1.0f + __expf(-x));
}

__device__ __forceinline__ v8f wmma_bf16(v16bf a, v16bf b, v8f c) {
    return __builtin_amdgcn_wmma_f32_16x16x32_bf16(false, a, false, b, (short)0, c,
                                                   false, false);
}

__global__ void gemm_kernel(const __bf16* __restrict__ Z, float* __restrict__ out) {
    int lane = threadIdx.x;                 // 0..31
    int wave = threadIdx.y;                 // 0..7
    int W = blockIdx.x * 8 + wave;          // 0..65535 wave-tiles
    int iTile = W >> 8;                     // 256 tiles of 32 rows
    int jTile = W & 255;
    int iBase = iTile * 32;
    int jBase = jTile * 32;
    int lm = lane & 15;
    int lh = lane >> 4;

    const __bf16* Ar0 = Z + (size_t)(iBase + lm)      * LDZ;
    const __bf16* Ar1 = Z + (size_t)(iBase + 16 + lm) * LDZ;
    const __bf16* Br0 = Z + (size_t)(jBase + lm)      * LDZ;
    const __bf16* Br1 = Z + (size_t)(jBase + 16 + lm) * LDZ;
    int kA = lh * 16;    // A lane layout: K = [kA..kA+7] then [kA+8..kA+15]
    int kB = lh * 16;    // B lane layout: K = [kB..kB+15] contiguous

    v8f c00 = {}, c01 = {}, c10 = {}, c11 = {};
#pragma unroll
    for (int k0 = 0; k0 < LDZ; k0 += 32) {
        v8bf a0lo = *(const v8bf*)(Ar0 + k0 + kA);
        v8bf a0hi = *(const v8bf*)(Ar0 + k0 + kA + 8);
        v8bf a1lo = *(const v8bf*)(Ar1 + k0 + kA);
        v8bf a1hi = *(const v8bf*)(Ar1 + k0 + kA + 8);
        v16bf a0 = __builtin_shufflevector(a0lo, a0hi, 0,1,2,3,4,5,6,7,8,9,10,11,12,13,14,15);
        v16bf a1 = __builtin_shufflevector(a1lo, a1hi, 0,1,2,3,4,5,6,7,8,9,10,11,12,13,14,15);
        v16bf b0 = *(const v16bf*)(Br0 + k0 + kB);
        v16bf b1 = *(const v16bf*)(Br1 + k0 + kB);
        c00 = wmma_bf16(a0, b0, c00);
        c01 = wmma_bf16(a0, b1, c01);
        c10 = wmma_bf16(a1, b0, c10);
        c11 = wmma_bf16(a1, b1, c11);
    }

    // C/D layout: VGPR r, lanes 0-15 -> M=r, lanes 16-31 -> M=8+r; N = lane%16
    int col0 = jBase + lm;
#pragma unroll
    for (int r = 0; r < 8; ++r) {
        int rowOff = r + lh * 8;
        float* o0 = out + (size_t)(iBase + rowOff) * N_NODES;
        float* o1 = out + (size_t)(iBase + 16 + rowOff) * N_NODES;
        o0[col0]      = sigmoidf(c00[r]);
        o0[col0 + 16] = sigmoidf(c01[r]);
        o1[col0]      = sigmoidf(c10[r]);
        o1[col0 + 16] = sigmoidf(c11[r]);
    }
}

// ---------------- launcher ----------------
extern "C" void kernel_launch(void* const* d_in, const int* in_sizes, int n_in,
                              void* d_out, int out_size, void* d_ws, size_t ws_size,
                              hipStream_t stream) {
    const float* x     = (const float*)d_in[0];
    const float* Wm    = (const float*)d_in[1];
    const float* Wl    = (const float*)d_in[2];
    const float* noise = (const float*)d_in[3];
    const int*   ei    = (const int*)d_in[4];
    // d_in[5] = num_nodes scalar (constant 8192), ignored
    (void)in_sizes; (void)n_in; (void)out_size; (void)ws_size;

    char* ws = (char*)d_ws;
    const size_t maskOff = 0;                                  // 8 MB
    const size_t hmOff   = maskOff + (size_t)MASK_WORDS * 4;   // 9.83 MB
    const size_t hlOff   = hmOff + (size_t)N_NODES * EMB * 4;  // 9.83 MB
    const size_t zOff    = hlOff + (size_t)N_NODES * EMB * 4;  // 5.24 MB bf16

    unsigned* mask = (unsigned*)(ws + maskOff);
    float*    Hm   = (float*)(ws + hmOff);
    float*    Hl   = (float*)(ws + hlOff);
    __bf16*   Z    = (__bf16*)(ws + zOff);
    float*    out  = (float*)d_out;

    // 1. zero bitmask: 2,097,152 words = 2048 blocks * 256 threads * uint4
    zero_mask_kernel<<<2048, 256, 0, stream>>>((uint4*)mask);
    // 2. scatter edges
    edge_kernel<<<N_EDGES / 256, 256, 0, stream>>>(ei, mask);
    // 3. Hm/Hl = x @ W (f32)
    xw_kernel<<<dim3(3, N_NODES / 8), 128, 0, stream>>>(x, Wm, Wl, Hm, Hl);
    // 4. sparse aggregate + reparameterize -> Z bf16
    agg_kernel<<<N_NODES, LDZ, 0, stream>>>(mask, Hm, Hl, noise, Z);
    // 5. sigmoid(Z @ Z^T) via v_wmma_f32_16x16x32_bf16
    gemm_kernel<<<8192, dim3(32, 8), 0, stream>>>(Z, out);
}
